// DeepseekMlp_13889924235944
// MI455X (gfx1250) — matrix operand order repaired and checked by reading, not compiled, not run on hardware
//
#include <hip/hip_runtime.h>
#include <hip/hip_bf16.h>

// ---------------------------------------------------------------------------
// DeepSeek MLP: out = (silu(x @ dq(w0,s0)^T) * (x @ dq(w1,s1)^T)) @ dq(w2,s2)^T
// bf16 WMMA with f32 accumulate; on-the-fly dequant (block scale folded into
// f32->bf16 convert); double-buffered LDS (1 barrier / K-step); async
// global->LDS copies for the bf16 activation tiles in the down-projection.
// ---------------------------------------------------------------------------

typedef __attribute__((ext_vector_type(16))) __bf16 v16bf;
typedef __attribute__((ext_vector_type(8)))  __bf16 v8bf;
typedef __attribute__((ext_vector_type(8)))  float  v8f;
typedef __attribute__((ext_vector_type(4)))  float  v4f;

#define DM   2048      // d_model
#define IN   5632      // intermediate
#define BM   128       // token rows per workgroup
#define BN   64        // weight rows (output cols) per workgroup
#define BK   32        // K step (one wmma K)
#define LSTR 40        // LDS row stride in bf16 elems (padded: conflict-free)

union Frag { v16bf v; v8bf h[2]; };

static __device__ inline v8bf cvt8(const float* f, float s) {
  v8bf r;
#pragma unroll
  for (int i = 0; i < 8; ++i) r[i] = (__bf16)(f[i] * s);
  return r;
}

static __device__ inline v8f wmma_bf16(const Frag& a, const Frag& b, v8f c) {
  return __builtin_amdgcn_wmma_f32_16x16x32_bf16(
      false, a.v, false, b.v, (short)0, c, false, false);
}

static __device__ inline float silu_mul(float g, float u) {
  const float e = __expf(-g);
#if __has_builtin(__builtin_amdgcn_rcpf)
  return g * __builtin_amdgcn_rcpf(1.0f + e) * u;
#else
  return g / (1.0f + e) * u;
#endif
}

// async global -> LDS copy of 32 contiguous bytes (two b128 beats).
// INST_OFFSET is added to BOTH the LDS and global addresses per the ISA.
static __device__ inline void async_copy32(unsigned lds_off, const void* gaddr) {
  asm volatile("global_load_async_to_lds_b128 %0, %1, off\n\t"
               "global_load_async_to_lds_b128 %0, %1, off offset:16"
               :: "v"(lds_off), "v"(gaddr) : "memory");
}
static __device__ inline void wait_async0() {
  asm volatile("s_wait_asynccnt 0x0" ::: "memory");
}

// ---------------------------------------------------------------------------
// Kernel 1: fused gate/up.  H[t, n] = silu(x@dw0^T) * (x@dw1^T)  (bf16 out)
// grid = (IN/BN, T/BM), block = 256 (8 waves: 4 along M x 2 along N)
// ---------------------------------------------------------------------------
__global__ __launch_bounds__(256)
void mlp_gate_up(const float* __restrict__ x,
                 const float* __restrict__ w0,
                 const float* __restrict__ s0,
                 const float* __restrict__ w1,
                 const float* __restrict__ s1,
                 unsigned short* __restrict__ Hraw)
{
  __shared__ __align__(16) __bf16 lx [2][BM * LSTR];
  __shared__ __align__(16) __bf16 lb0[2][BN * LSTR];
  __shared__ __align__(16) __bf16 lb1[2][BN * LSTR];
  __bf16* H = (__bf16*)Hraw;

  const int tid  = threadIdx.x;
  const int lane = tid & 31;
  const int wave = tid >> 5;
  const int wm   = wave & 3;            // M sub-tile: wm*32
  const int wn   = wave >> 2;           // N sub-tile: wn*32

  const int mBase = blockIdx.y * BM;
  const int nBase = blockIdx.x * BN;

  const int xr = tid >> 1;              // staging: x rows
  const int xc = (tid & 1) * 16;
  const int wr = tid >> 2;              // staging: weight rows
  const int wc = (tid & 3) * 8;

  const int fRow = lane & 15;           // fragment coords (contiguous k = 16*(lane/16) + s, A and B)
  const int fKa  = (lane >> 4) * 16;
  const int fKb  = (lane >> 4) * 16;

  const int sRow = (nBase >> 7) * (DM >> 7);

  const float* xp0 = x  + (size_t)(mBase + xr) * DM + xc;
  const float* wp0 = w0 + (size_t)(nBase + wr) * DM + wc;
  const float* wp1 = w1 + (size_t)(nBase + wr) * DM + wc;

  v8f accG[2][2] = {};
  v8f accU[2][2] = {};

  float xf[16], wf0[8], wf1[8], sc0, sc1;

  auto glb = [&](int k) {
    sc0 = s0[sRow + (k >> 7)];
    sc1 = s1[sRow + (k >> 7)];
    *(v4f*)(xf + 0)  = *(const v4f*)(xp0 + k + 0);
    *(v4f*)(xf + 4)  = *(const v4f*)(xp0 + k + 4);
    *(v4f*)(xf + 8)  = *(const v4f*)(xp0 + k + 8);
    *(v4f*)(xf + 12) = *(const v4f*)(xp0 + k + 12);
    *(v4f*)(wf0 + 0) = *(const v4f*)(wp0 + k + 0);
    *(v4f*)(wf0 + 4) = *(const v4f*)(wp0 + k + 4);
    *(v4f*)(wf1 + 0) = *(const v4f*)(wp1 + k + 0);
    *(v4f*)(wf1 + 4) = *(const v4f*)(wp1 + k + 4);
  };
  auto stash = [&](int b) {
    *(v8bf*)&lx [b][xr * LSTR + xc + 0] = cvt8(xf + 0, 1.0f);
    *(v8bf*)&lx [b][xr * LSTR + xc + 8] = cvt8(xf + 8, 1.0f);
    *(v8bf*)&lb0[b][wr * LSTR + wc]     = cvt8(wf0, sc0);
    *(v8bf*)&lb1[b][wr * LSTR + wc]     = cvt8(wf1, sc1);
  };

  glb(0);
  stash(0);
  __syncthreads();

  int p = 0;
  for (int k0 = 0; k0 < DM; k0 += BK) {
    const int kn  = (k0 + BK < DM) ? k0 + BK : 0;        // next tile (clamped)
    const int kpf = (kn + BK < DM) ? kn + BK : 0;        // prefetch 2 ahead
    glb(kn);
    __builtin_prefetch(wp0 + kpf, 0, 1);
    __builtin_prefetch(wp1 + kpf, 0, 1);

    Frag a[2], b0[2], b1[2];
#pragma unroll
    for (int mt = 0; mt < 2; ++mt) {
      const __bf16* q = &lx[p][(wm * 32 + mt * 16 + fRow) * LSTR + fKa];
      a[mt].h[0] = *(const v8bf*)(q);
      a[mt].h[1] = *(const v8bf*)(q + 8);
    }
#pragma unroll
    for (int nt = 0; nt < 2; ++nt) {
      const __bf16* q0 = &lb0[p][(wn * 32 + nt * 16 + fRow) * LSTR + fKb];
      b0[nt].h[0] = *(const v8bf*)(q0);
      b0[nt].h[1] = *(const v8bf*)(q0 + 8);
      const __bf16* q1 = &lb1[p][(wn * 32 + nt * 16 + fRow) * LSTR + fKb];
      b1[nt].h[0] = *(const v8bf*)(q1);
      b1[nt].h[1] = *(const v8bf*)(q1 + 8);
    }
#pragma unroll
    for (int mt = 0; mt < 2; ++mt)
#pragma unroll
      for (int nt = 0; nt < 2; ++nt) {
        accG[mt][nt] = wmma_bf16(a[mt], b0[nt], accG[mt][nt]);
        accU[mt][nt] = wmma_bf16(a[mt], b1[nt], accU[mt][nt]);
      }

    stash(p ^ 1);          // writes the *other* buffer: no hazard with reads
    __syncthreads();       // one barrier per K-step
    p ^= 1;
  }

  const int rowHalf = (lane >> 4) * 8;
  const int colOff  = lane & 15;
#pragma unroll
  for (int mt = 0; mt < 2; ++mt)
#pragma unroll
    for (int nt = 0; nt < 2; ++nt) {
      const int col = nBase + wn * 32 + nt * 16 + colOff;
#pragma unroll
      for (int r = 0; r < 8; ++r) {
        const int row = mBase + wm * 32 + mt * 16 + rowHalf + r;
        H[(size_t)row * IN + col] =
            (__bf16)silu_mul(accG[mt][nt][r], accU[mt][nt][r]);
      }
    }
}

// ---------------------------------------------------------------------------
// Kernel 2: out[t, d] = H @ dq(w2,s2)^T   (f32 out)
// A (bf16) tiles move global->LDS via async copies (ASYNCcnt).
// grid = (DM/BN, T/BM), block = 256
// ---------------------------------------------------------------------------
__global__ __launch_bounds__(256)
void mlp_down(const unsigned short* __restrict__ Hraw,
              const float* __restrict__ w2,
              const float* __restrict__ s2,
              float* __restrict__ out)
{
  __shared__ __align__(16) __bf16 la[2][BM * LSTR];
  __shared__ __align__(16) __bf16 lb[2][BN * LSTR];
  const __bf16* H = (const __bf16*)Hraw;

  const int tid  = threadIdx.x;
  const int lane = tid & 31;
  const int wave = tid >> 5;
  const int wm   = wave & 3;
  const int wn   = wave >> 2;

  const int mBase = blockIdx.y * BM;
  const int nBase = blockIdx.x * BN;

  const int ar = tid >> 1;              // A staging rows (32B per thread)
  const int ac = (tid & 1) * 16;
  const int wr = tid >> 2;
  const int wc = (tid & 3) * 8;

  const int fRow = lane & 15;
  const int fKa  = (lane >> 4) * 16;
  const int fKb  = (lane >> 4) * 16;

  const int sRow = (nBase >> 7) * (IN >> 7);

  const __bf16* hp0 = H  + (size_t)(mBase + ar) * IN + ac;
  const float*  wp  = w2 + (size_t)(nBase + wr) * IN + wc;

  v8f acc[2][2] = {};
  float wf[8], sc;

  auto glbw = [&](int k) {
    sc = s2[sRow + (k >> 7)];
    *(v4f*)(wf + 0) = *(const v4f*)(wp + k + 0);
    *(v4f*)(wf + 4) = *(const v4f*)(wp + k + 4);
  };
  auto asyncA = [&](int k, int b) {
    // flat pointer to __shared__ carries the LDS offset in its low 32 bits
    const unsigned loff =
        (unsigned)(unsigned long long)(uintptr_t)&la[b][ar * LSTR + ac];
    async_copy32(loff, hp0 + k);
  };
  auto stashw = [&](int b) {
    *(v8bf*)&lb[b][wr * LSTR + wc] = cvt8(wf, sc);
  };

  asyncA(0, 0);
  glbw(0);
  stashw(0);
  wait_async0();
  __syncthreads();

  int p = 0;
  for (int k0 = 0; k0 < IN; k0 += BK) {
    const int kn  = (k0 + BK < IN) ? k0 + BK : 0;
    const int kpf = (kn + BK < IN) ? kn + BK : 0;
    glbw(kn);
    asyncA(kn, p ^ 1);                 // overlap DMA with this tile's WMMAs
    __builtin_prefetch(wp + kpf, 0, 1);

    Frag a[2], b[2];
#pragma unroll
    for (int mt = 0; mt < 2; ++mt) {
      const __bf16* q = &la[p][(wm * 32 + mt * 16 + fRow) * LSTR + fKa];
      a[mt].h[0] = *(const v8bf*)(q);
      a[mt].h[1] = *(const v8bf*)(q + 8);
    }
#pragma unroll
    for (int nt = 0; nt < 2; ++nt) {
      const __bf16* q = &lb[p][(wn * 32 + nt * 16 + fRow) * LSTR + fKb];
      b[nt].h[0] = *(const v8bf*)(q);
      b[nt].h[1] = *(const v8bf*)(q + 8);
    }
#pragma unroll
    for (int mt = 0; mt < 2; ++mt)
#pragma unroll
      for (int nt = 0; nt < 2; ++nt)
        acc[mt][nt] = wmma_bf16(a[mt], b[nt], acc[mt][nt]);

    stashw(p ^ 1);
    wait_async0();                     // async tile for next iter in place
    __syncthreads();
    p ^= 1;
  }

  const int rowHalf = (lane >> 4) * 8;
  const int colOff  = lane & 15;
#pragma unroll
  for (int mt = 0; mt < 2; ++mt)
#pragma unroll
    for (int nt = 0; nt < 2; ++nt) {
      const int col = nBase + wn * 32 + nt * 16 + colOff;
#pragma unroll
      for (int r = 0; r < 8; ++r) {
        const int row = mBase + wm * 32 + mt * 16 + rowHalf + r;
        out[(size_t)row * DM + col] = acc[mt][nt][r];
      }
    }
}

// ---------------------------------------------------------------------------
extern "C" void kernel_launch(void* const* d_in, const int* in_sizes, int n_in,
                              void* d_out, int out_size, void* d_ws, size_t ws_size,
                              hipStream_t stream) {
  (void)n_in; (void)out_size; (void)ws_size;
  const float* x  = (const float*)d_in[0];
  const float* w0 = (const float*)d_in[1];
  const float* s0 = (const float*)d_in[2];
  const float* w1 = (const float*)d_in[3];
  const float* s1 = (const float*)d_in[4];
  const float* w2 = (const float*)d_in[5];
  const float* s2 = (const float*)d_in[6];
  float* out = (float*)d_out;
  unsigned short* H = (unsigned short*)d_ws;   // bf16 activations [T, IN]

  const int T = in_sizes[0] / DM;              // tokens (8192)

  dim3 g1(IN / BN, T / BM);                    // 88 x 64
  mlp_gate_up<<<g1, 256, 0, stream>>>(x, w0, s0, w1, s1, H);

  dim3 g2(DM / BN, T / BM);                    // 32 x 64
  mlp_down<<<g2, 256, 0, stream>>>(H, w2, s2, out);
}
